// MolecularGCN_53755810677401
// MI455X (gfx1250) — matrix operand order repaired and checked by reading, not compiled, not run on hardware
//
#include <hip/hip_runtime.h>
#include <hip/hip_bf16.h>
#include <stdint.h>

#define NN 100000
#define NNP 100096            // NN padded to multiple of 128
#define NE 3200000
#define NF 128
#define GD 64
#define NG 2048
#define BN_EPS 1e-5f
#define KEEP 0.9f

typedef __attribute__((ext_vector_type(16))) _Float16 v16h;
typedef __attribute__((ext_vector_type(8)))  float    v8f;

// ---------------------------------------------------------------------------
// CDNA5 async global->LDS copy (16B per lane), tracked by ASYNCcnt.
// LDS address = low 32 bits of the generic shared-memory address (ISA 10.2).
// ---------------------------------------------------------------------------
__device__ __forceinline__ void async_copy_b128(void* lds_ptr, const void* gptr) {
  uint32_t lds_off = (uint32_t)(uintptr_t)lds_ptr;
  asm volatile("global_load_async_to_lds_b128 %0, %1, off"
               :: "v"(lds_off), "v"(gptr) : "memory");
}
__device__ __forceinline__ void wait_async0() {
  asm volatile("s_wait_asynccnt 0x0" ::: "memory");
}

// ---------------------------------------------------------------------------
// WMMA GEMM on pre-packed f16 inputs, K compile-time (64 or 128):
//   Y[m][n] = relu?( Xh[m][:] @ Wh[n][:] + bias[n] )
// Xh:[rows_pad][K] f16 (pad rows zeroed), Wh:[64][K] f16.
// 256 threads = 8 waves; each wave owns a 16x64 output tile.
// Whole A tile (128xK) + whole W staged to LDS via async DMA once;
// single wait+barrier, then a fully unrolled ds_load+WMMA loop.
// ---------------------------------------------------------------------------
template <int K>
__global__ __launch_bounds__(256) void k_wmma_gemm16(
    const _Float16* __restrict__ Xh, int nrows,
    const _Float16* __restrict__ Wh,
    const float* __restrict__ bias, int do_relu,
    float* __restrict__ Y)
{
  constexpr int ST = K + 8;           // LDS row stride in halves (bank-safe)
  constexpr int KC = K / 8;           // 16B chunks per row
  __shared__ _Float16 Wl[64 * ST];
  __shared__ _Float16 Al[128 * ST];

  const int tid  = threadIdx.x;
  const int wave = tid >> 5;
  const int lane = tid & 31;
  const int half = lane >> 4;
  const int l16  = lane & 15;
  const int row0 = blockIdx.x * 128;

  // async-stage all weights (64 x K) and the whole A tile (128 x K)
  {
    const int r = tid / KC, c = (tid % KC) * 8;
    #pragma unroll
    for (int t = 0; t < (64 * KC) / 256; ++t) {
      const int rr = r + t * (256 / KC);
      async_copy_b128(&Wl[rr * ST + c], Wh + (size_t)rr * K + c);
    }
    #pragma unroll
    for (int t = 0; t < (128 * KC) / 256; ++t) {
      const int rr = r + t * (256 / KC);
      async_copy_b128(&Al[rr * ST + c], Xh + (size_t)(row0 + rr) * K + c);
    }
  }
  wait_async0();
  __syncthreads();

  v8f acc[4];
  #pragma unroll
  for (int t = 0; t < 4; ++t) acc[t] = (v8f){0.f,0.f,0.f,0.f,0.f,0.f,0.f,0.f};

  #pragma unroll
  for (int k0 = 0; k0 < K; k0 += 32) {
    // A fragment: 16x32 f16, k = 16*half + s (one contiguous 32B run per lane)
    v16h a;
    {
      const int mbase = (wave * 16 + l16) * ST + k0;
      #pragma unroll
      for (int j = 0; j < 8; ++j) {
        const int koff = 16 * half + 2 * j;
        a[2 * j]     = Al[mbase + koff];
        a[2 * j + 1] = Al[mbase + koff + 1];
      }
    }
    // B fragments (B[k][n] = W[n][k]) + WMMA over the 4 16-col tiles
    #pragma unroll
    for (int nt = 0; nt < 4; ++nt) {
      const int base = (nt * 16 + l16) * ST + k0 + 16 * half;
      v16h b;
      #pragma unroll
      for (int e = 0; e < 16; ++e) b[e] = Wl[base + e];
      acc[nt] = __builtin_amdgcn_wmma_f32_16x16x32_f16(
          false, a, false, b, (short)0, acc[nt], false, false);
    }
  }

  #pragma unroll
  for (int nt = 0; nt < 4; ++nt) {
    const int n = nt * 16 + l16;
    #pragma unroll
    for (int r = 0; r < 8; ++r) {
      const int m = row0 + wave * 16 + r + 8 * half;
      if (m < nrows) {
        float v = acc[nt][r];
        if (bias) v += bias[n];
        if (do_relu) v = fmaxf(v, 0.f);
        Y[(size_t)m * 64 + n] = v;
      }
    }
  }
}

// ---------------------------------------------------------------------------
// f32 -> f16 pack (4 elements/thread) with optional per-column affine
// (fused BN scale/shift) and zero fill of padding rows.
// ---------------------------------------------------------------------------
__global__ void k_cvt_f16(const float* __restrict__ X,
                          const float* __restrict__ cs, const float* __restrict__ cb,
                          int rows, int rows_pad, int cols,
                          _Float16* __restrict__ O) {
  int i4 = blockIdx.x * 256 + threadIdx.x;
  const int total = (rows_pad * cols) >> 2;
  if (i4 < total) {
    const int i = i4 << 2;
    const int r = i / cols, c0 = i % cols;
    float4 v = make_float4(0.f, 0.f, 0.f, 0.f);
    if (r < rows) {
      v = *(const float4*)(X + (size_t)r * cols + c0);
      if (cs) {
        v.x = v.x * cs[c0]     + cb[c0];
        v.y = v.y * cs[c0 + 1] + cb[c0 + 1];
        v.z = v.z * cs[c0 + 2] + cb[c0 + 2];
        v.w = v.w * cs[c0 + 3] + cb[c0 + 3];
      }
    }
    O[i]     = (_Float16)v.x;
    O[i + 1] = (_Float16)v.y;
    O[i + 2] = (_Float16)v.z;
    O[i + 3] = (_Float16)v.w;
  }
}

// ---------------------------------------------------------------------------
__global__ void k_zero_f(float* p, int n) {
  int i = blockIdx.x * blockDim.x + threadIdx.x;
  if (i < n) p[i] = 0.f;
}

__global__ void k_deg(const int* __restrict__ dst, float* deg, int ne) {
  int i = blockIdx.x * blockDim.x + threadIdx.x;
  if (i < ne) atomicAdd(&deg[dst[i]], 1.0f);
}

__global__ void k_rsqrt(float* d, int n) {
  int i = blockIdx.x * blockDim.x + threadIdx.x;
  if (i < n) d[i] = rsqrtf(d[i] + 1.0f);
}

// edge scatter: 16 threads/edge, float4 gather + 4 L2-resident f32 atomics
__global__ void k_edge(const int* __restrict__ src, const int* __restrict__ dst,
                       const float* __restrict__ dis, const float* __restrict__ T,
                       float* __restrict__ agg) {
  int idx = blockIdx.x * 256 + threadIdx.x;
  if (idx < NE * 16) {
    const int e = idx >> 4, q = idx & 15;
    const int s = src[e], d = dst[e];
    const float c = dis[s] * dis[d];
    const float4 t = ((const float4*)T)[(size_t)s * 16 + q];
    float* ap = agg + (size_t)d * 64 + q * 4;
    atomicAdd(ap + 0, t.x * c);
    atomicAdd(ap + 1, t.y * c);
    atomicAdd(ap + 2, t.z * c);
    atomicAdd(ap + 3, t.w * c);
  }
}

// self-loop term + bias + relu, fused column sum/sumsq for training-mode BN
__global__ __launch_bounds__(256) void k_finalize(
    const float* __restrict__ T, const float* __restrict__ agg,
    const float* __restrict__ dis, const float* __restrict__ bias,
    float* __restrict__ H, float* __restrict__ sums, int nrows)
{
  __shared__ float sbuf[512];
  const int f = threadIdx.x & 63, rg = threadIdx.x >> 6;
  const int n0 = blockIdx.x * 256;
  float s = 0.f, s2 = 0.f;
  const float b = bias[f];
  for (int n = n0 + rg; n < n0 + 256; n += 4) {
    if (n < nrows) {
      const float di = dis[n];
      float v = T[(size_t)n * 64 + f] * di * di + agg[(size_t)n * 64 + f] + b;
      v = fmaxf(v, 0.f);
      H[(size_t)n * 64 + f] = v;
      s += v; s2 += v * v;
    }
  }
  sbuf[threadIdx.x] = s; sbuf[256 + threadIdx.x] = s2;
  __syncthreads();
  if (rg == 0) {
    float ts  = sbuf[f] + sbuf[64+f] + sbuf[128+f] + sbuf[192+f];
    float ts2 = sbuf[256+f] + sbuf[320+f] + sbuf[384+f] + sbuf[448+f];
    atomicAdd(&sums[f], ts);
    atomicAdd(&sums[64 + f], ts2);
  }
}

__global__ void k_scaleshift(const float* __restrict__ sums,
                             const float* __restrict__ gamma,
                             const float* __restrict__ beta,
                             float invN, float* cs, float* cb) {
  const int f = threadIdx.x;  // 64
  const float mean = sums[f] * invN;
  const float var  = sums[64 + f] * invN - mean * mean;
  const float sc   = gamma[f] * rsqrtf(var + BN_EPS);
  cs[f] = sc;
  cb[f] = beta[f] - mean * sc;
}

// global add-pool with layer-3 BN folded in
__global__ void k_pool(const float* __restrict__ H, const float* __restrict__ cs,
                       const float* __restrict__ cb, const int* __restrict__ batch,
                       float* __restrict__ g, int nrows) {
  int idx = blockIdx.x * 256 + threadIdx.x;
  if (idx < nrows * 64) {
    const int n = idx >> 6, f = idx & 63;
    const float v = H[idx] * cs[f] + cb[f];
    atomicAdd(&g[(size_t)batch[n] * 64 + f], v);
  }
}

// column sums/sumsq over pooled matrix (one block per column)
__global__ __launch_bounds__(256) void k_colstats(const float* __restrict__ X,
                                                  int rows, float* __restrict__ sums) {
  const int f = blockIdx.x;
  float s = 0.f, s2 = 0.f;
  for (int r = threadIdx.x; r < rows; r += 256) {
    const float v = X[(size_t)r * 64 + f];
    s += v; s2 += v * v;
  }
  __shared__ float b1[256], b2[256];
  b1[threadIdx.x] = s; b2[threadIdx.x] = s2;
  __syncthreads();
  for (int st = 128; st > 0; st >>= 1) {
    if (threadIdx.x < st) { b1[threadIdx.x] += b1[threadIdx.x + st];
                            b2[threadIdx.x] += b2[threadIdx.x + st]; }
    __syncthreads();
  }
  if (threadIdx.x == 0) { sums[f] = b1[0]; sums[64 + f] = b2[0]; }
}

__device__ inline float u01hash(uint32_t a, uint32_t b) {
  uint32_t h = a * 0x9E3779B9u ^ (b + 0x7F4A7C15u) * 0x85EBCA6Bu;
  h ^= h >> 16; h *= 0x7FEB352Du; h ^= h >> 15; h *= 0x846CA68Bu; h ^= h >> 16;
  return (float)(h >> 8) * (1.0f / 16777216.0f);
}

// BN + dropout; also emits the next GEMM's f16 input
__global__ void k_bn_dropout(const float* __restrict__ Z, const float* __restrict__ cs,
                             const float* __restrict__ cb, int layer,
                             float* __restrict__ G, _Float16* __restrict__ G16) {
  int idx = blockIdx.x * 256 + threadIdx.x;
  if (idx < NG * 64) {
    const int f = idx & 63;
    float v = Z[idx] * cs[f] + cb[f];
    const float r = u01hash((uint32_t)layer, (uint32_t)idx);
    v = (r < KEEP) ? v * (1.0f / KEEP) : 0.f;
    G[idx]   = v;
    G16[idx] = (_Float16)v;
  }
}

__global__ void k_outproj(const float* __restrict__ G, const float* __restrict__ W,
                          const float* __restrict__ b, float* __restrict__ out) {
  int r = blockIdx.x * blockDim.x + threadIdx.x;
  if (r < NG) {
    float s = b[0];
    #pragma unroll 8
    for (int k = 0; k < 64; ++k) s += G[(size_t)r * 64 + k] * W[k];
    out[r] = s;
  }
}

// ---------------------------------------------------------------------------
extern "C" void kernel_launch(void* const* d_in, const int* in_sizes, int n_in,
                              void* d_out, int out_size, void* d_ws, size_t ws_size,
                              hipStream_t stream) {
  (void)in_sizes; (void)n_in; (void)out_size; (void)ws_size;

  const float* x        = (const float*)d_in[0];
  const float* convW[3] = {(const float*)d_in[1], (const float*)d_in[3], (const float*)d_in[5]};
  const float* convB[3] = {(const float*)d_in[2], (const float*)d_in[4], (const float*)d_in[6]};
  const float* bng = (const float*)d_in[7];
  const float* bnb = (const float*)d_in[8];
  const float* dW  = (const float*)d_in[9];
  const float* dB  = (const float*)d_in[10];
  const float* dng = (const float*)d_in[11];
  const float* dnb = (const float*)d_in[12];
  const float* oW  = (const float*)d_in[13];
  const float* oB  = (const float*)d_in[14];
  const int*   ei  = (const int*)d_in[15];
  const int*   bat = (const int*)d_in[16];
  const int* src = ei;
  const int* dst = ei + NE;

  float* ws = (float*)d_ws;
  size_t o = 0;
  float* dis  = ws + o; o += 100352;
  float* T    = ws + o; o += (size_t)NN * 64;
  float* agg  = ws + o; o += (size_t)NN * 64;
  float* H    = ws + o; o += (size_t)NN * 64;
  float* g0   = ws + o; o += (size_t)NG * 64;
  float* g1   = ws + o; o += (size_t)NG * 64;
  float* sums = ws + o; o += 6 * 128;             // per-layer [sum|sumsq]
  float* scal = ws + o; o += 6 * 128;             // per-layer [cs|cb]
  _Float16* Ah16 = (_Float16*)(ws + o); o += (size_t)NNP * 128 / 2;  // f16 GEMM input
  _Float16* Wh16 = (_Float16*)(ws + o); o += 64 * 128 / 2;           // f16 weights

  auto cdiv = [](long long a, long long b) { return (int)((a + b - 1) / b); };

  // degree^{-1/2} with self-loops
  k_zero_f<<<cdiv(NN, 256), 256, 0, stream>>>(dis, NN);
  k_deg<<<cdiv(NE, 256), 256, 0, stream>>>(dst, dis, NE);
  k_rsqrt<<<cdiv(NN, 256), 256, 0, stream>>>(dis, NN);

  // three GCN conv layers
  for (int i = 0; i < 3; ++i) {
    const int K = (i == 0) ? NF : GD;
    const float* X  = (i == 0) ? x : H;
    const float* cs = (i == 0) ? nullptr : scal + (i - 1) * 128;
    const float* cb = (i == 0) ? nullptr : scal + (i - 1) * 128 + 64;
    // pack inputs/weights to f16 (previous layer's BN fused into the pack)
    k_cvt_f16<<<cdiv((long long)NNP * K / 4, 256), 256, 0, stream>>>(
        X, cs, cb, NN, NNP, K, Ah16);
    k_cvt_f16<<<cdiv(64LL * K / 4, 256), 256, 0, stream>>>(
        convW[i], nullptr, nullptr, 64, 64, K, Wh16);
    if (K == NF)
      k_wmma_gemm16<NF><<<NNP / 128, 256, 0, stream>>>(Ah16, NN, Wh16,
                                                       nullptr, 0, T);
    else
      k_wmma_gemm16<GD><<<NNP / 128, 256, 0, stream>>>(Ah16, NN, Wh16,
                                                       nullptr, 0, T);
    k_zero_f<<<cdiv((long long)NN * 64, 256), 256, 0, stream>>>(agg, NN * 64);
    k_zero_f<<<1, 128, 0, stream>>>(sums + i * 128, 128);
    k_edge<<<cdiv((long long)NE * 16, 256), 256, 0, stream>>>(src, dst, dis, T, agg);
    k_finalize<<<cdiv(NN, 256), 256, 0, stream>>>(T, agg, dis, convB[i], H,
                                                  sums + i * 128, NN);
    k_scaleshift<<<1, 64, 0, stream>>>(sums + i * 128, bng + i * 64, bnb + i * 64,
                                       1.0f / NN, scal + i * 128, scal + i * 128 + 64);
  }

  // global add-pool with BN3 folded in
  k_zero_f<<<cdiv(NG * 64, 256), 256, 0, stream>>>(g0, NG * 64);
  k_pool<<<cdiv((long long)NN * 64, 256), 256, 0, stream>>>(
      H, scal + 2 * 128, scal + 2 * 128 + 64, bat, g0, NN);

  // dense head (layer-0 f16 input from a pack; later layers from k_bn_dropout)
  k_cvt_f16<<<cdiv(NG * 64 / 4, 256), 256, 0, stream>>>(
      g0, nullptr, nullptr, NG, NG, 64, Ah16);
  float* gin = g0; float* gout = g1;
  for (int i = 0; i < 3; ++i) {
    k_cvt_f16<<<cdiv(64 * 64 / 4, 256), 256, 0, stream>>>(
        dW + i * 4096, nullptr, nullptr, 64, 64, 64, Wh16);
    k_wmma_gemm16<GD><<<NG / 128, 256, 0, stream>>>(Ah16, NG, Wh16,
                                                    dB + i * 64, 1, T);
    k_colstats<<<64, 256, 0, stream>>>(T, NG, sums + (3 + i) * 128);
    k_scaleshift<<<1, 64, 0, stream>>>(sums + (3 + i) * 128, dng + i * 64, dnb + i * 64,
                                       1.0f / NG, scal + (3 + i) * 128,
                                       scal + (3 + i) * 128 + 64);
    k_bn_dropout<<<cdiv(NG * 64, 256), 256, 0, stream>>>(
        T, scal + (3 + i) * 128, scal + (3 + i) * 128 + 64, i, gout, Ah16);
    float* t2 = gin; gin = gout; gout = t2;
  }

  k_outproj<<<cdiv(NG, 256), 256, 0, stream>>>(gin, oW, oB, (float*)d_out);
}
